// TemporalHeteroGNN_33449205301475
// MI455X (gfx1250) — matrix operand order repaired and checked by reading, not compiled, not run on hardware
//
#include <hip/hip_runtime.h>
#include <hip/hip_bf16.h>
#include <math.h>

// ---------------------------------------------------------------------------
// Problem constants (from reference)
// ---------------------------------------------------------------------------
#define C_IN   16
#define HID    128
#define OUTD   64
#define T_STEPS 24
#define F1     (C_IN + HID)   // 144
#define N_WS   5000
#define N_HRU  200000
#define N_CH   50000
#define N_GW   300000
#define N_TOT  (N_WS + N_HRU + N_CH + N_GW)  // 555000

// row offsets of each node type inside the packed per-node buffers
#define OFF_WS  0
#define OFF_HRU (N_WS)
#define OFF_CH  (N_WS + N_HRU)
#define OFF_GW  (N_WS + N_HRU + N_CH)

// LDS strides (in halves). 168 halves = 84 dwords, 84 % 64 == 20 -> the 16
// rows of an A-fragment land on 16 distinct 4-bank groups (optimal 2-pass
// ds_load_b128). 424 halves = 212 dwords (212%64==20), 296 halves = 148
// dwords (148%64==20): same property.
#define GRU_STRIDE 168
#define SAGE_STRIDE_MAX 424

typedef __attribute__((ext_vector_type(16))) _Float16 v16h;
typedef __attribute__((ext_vector_type(8)))  _Float16 v8h;
typedef __attribute__((ext_vector_type(8)))  float    v8f;

#define DEV static __device__ __forceinline__

// ---------------------------------------------------------------------------
// WMMA fragment helpers (CDNA5 16x16x32 f16, wave32)
//
// A (16x32, f16): lane L holds row M=L&15; halves e=0..7 -> K = (L>=16?8:0)+e,
//                 e=8..15 -> K = 16 + (L>=16?8:0) + (e-8).   (ISA 7.12.2)
// B (32x16, f16): lane L holds col N=L&15, halves e -> K as for A (pre-swizzled).
// C/D (16x16, f32): VGPR i: lanes0-15 -> (M=i,   N=lane),
//                            lanes16-31 -> (M=i+8, N=lane-16).
// ---------------------------------------------------------------------------
DEV v16h load_a_frag(const _Float16* row0, int strideH, int kbase) {
  const int lane = threadIdx.x & 31;
  const _Float16* p = row0 + (lane & 15) * strideH + kbase + ((lane >> 4) << 3);
  v8h lo = *(const v8h*)(p);
  v8h hi = *(const v8h*)(p + 16);
  v16h a;
#pragma unroll
  for (int e = 0; e < 8; ++e) { a[e] = lo[e]; a[8 + e] = hi[e]; }
  return a;
}

DEV v16h load_b_frag(const _Float16* wf, int chunk, int ntiles, int nt) {
  const int lane = threadIdx.x & 31;
  const _Float16* p = wf + ((((size_t)chunk * ntiles + nt) * 32) + lane) * 16;
  return *(const v16h*)p;  // 32B contiguous -> 2x global_load_b128
}

DEV v8f wmma_f16(v16h a, v16h b, v8f c) {
  return __builtin_amdgcn_wmma_f32_16x16x32_f16(
      /*neg_a=*/false, a, /*neg_b=*/false, b,
      /*c_mod=*/(short)0, c, /*reuse_a=*/false, /*reuse_b=*/false);
}

// Fast transcendentals: v_exp_f32 + v_rcp_f32, no IEEE divide chains.
DEV float sigmoid_fast(float x) {
  return __builtin_amdgcn_rcpf(1.f + __expf(-x));
}
DEV float tanh_fast(float x) {
  // tanh(x) = 1 - 2/(exp(2x)+1)
  return 1.f - 2.f * __builtin_amdgcn_rcpf(1.f + __expf(2.f * x));
}

// ---------------------------------------------------------------------------
// Weight pre-swizzle: f32 W[K][O] row-major  ->  f16 B-fragment layout:
//   wf[((c*ntiles + nt)*32 + lane)*16 + e] = W[c*32 + 16*(e>>3) + 8*(lane>>4) + (e&7)][nt*16 + (lane&15)]
// K zero-padded to cL*32.
// ---------------------------------------------------------------------------
__global__ void wprep_kernel(const float* __restrict__ W, _Float16* __restrict__ wf,
                             int K, int O, int cL) {
  const int ntiles = O >> 4;
  const int total = cL * ntiles * 32 * 16;
  const int tid = blockIdx.x * blockDim.x + threadIdx.x;
  if (tid >= total) return;
  const int e = tid & 15;
  const int lane = (tid >> 4) & 31;
  const int rest = tid >> 9;          // c*ntiles + nt
  const int nt = rest % ntiles;
  const int c = rest / ntiles;
  const int k = c * 32 + ((e >> 3) << 4) + ((lane >> 4) << 3) + (e & 7);
  const int n = nt * 16 + (lane & 15);
  wf[tid] = (k < K) ? (_Float16)W[(size_t)k * O + n] : (_Float16)0;
}

// ---------------------------------------------------------------------------
// GRU: one block = 16 nodes, 8 waves = 8 N-tiles of HID=128.
// h kept register-resident in C-layout; f16 copies of [x|h] / [x|r*h] in LDS.
// B-fragments (15 x v16h) are reloaded from global at every step of the time loop.
// Writes feat[node][0:16]=x_static, feat[node][16:144]=h_final.
// ---------------------------------------------------------------------------
__global__ __launch_bounds__(256) void gru_kernel(
    const float* __restrict__ ts, const float* __restrict__ xs,
    const _Float16* __restrict__ wfz, const _Float16* __restrict__ wfr,
    const _Float16* __restrict__ wfh,
    const float* __restrict__ bz, const float* __restrict__ br,
    const float* __restrict__ bh,
    float* __restrict__ feat, int N) {
  __shared__ _Float16 xh[16 * GRU_STRIDE];
  __shared__ _Float16 xh2[16 * GRU_STRIDE];

  const int tid = threadIdx.x;
  const int lane = tid & 31;
  const int nt = tid >> 5;          // wave id == N-tile id
  const int nl = lane & 15;
  const int n = nt * 16 + nl;       // global output column (0..127)
  const int mbase = (lane >> 4) * 8;
  const int tile0 = blockIdx.x * 16;

  for (int i = tid; i < 16 * GRU_STRIDE; i += 256) {
    xh[i] = (_Float16)0; xh2[i] = (_Float16)0;   // K-pad (144..159) stays 0
  }

  // per-thread streaming pointer into ts[t][node][c]
  const int sr = tid >> 4, sc = tid & 15;
  int snode = tile0 + sr; if (snode >= N) snode = N - 1;
  const float* tsp = ts + (size_t)snode * C_IN + sc;
  const size_t tstep = (size_t)N * C_IN;
  const int ldsoff = sr * GRU_STRIDE + sc;

  const float bzv = bz[n], brv = br[n], bhv = bh[n];
  float h[8];
#pragma unroll
  for (int i = 0; i < 8; ++i) h[i] = 0.f;
  __syncthreads();

  for (int t = 0; t < T_STEPS; ++t) {
    {  // stage x_t (16 nodes x 16 channels), one element per thread
      const _Float16 v = (_Float16)(*tsp);
      __builtin_prefetch(tsp + tstep, 0, 1);   // global_prefetch_b8 next step
      tsp += tstep;
      xh[ldsoff] = v;
      xh2[ldsoff] = v;
    }
    __syncthreads();

    v8f az, ar;
#pragma unroll
    for (int i = 0; i < 8; ++i) { az[i] = bzv; ar[i] = brv; }
#pragma unroll
    for (int c = 0; c < 5; ++c) {   // K = 144 -> 5 chunks of 32 (padded)
      v16h a = load_a_frag(xh, GRU_STRIDE, c * 32);
      az = wmma_f16(a, load_b_frag(wfz, c, 8, nt), az);
      ar = wmma_f16(a, load_b_frag(wfr, c, 8, nt), ar);
    }
    float z[8];
#pragma unroll
    for (int i = 0; i < 8; ++i) {
      z[i] = sigmoid_fast(az[i]);
      float rr = sigmoid_fast(ar[i]);
      xh2[(mbase + i) * GRU_STRIDE + C_IN + n] = (_Float16)(rr * h[i]);
    }
    __syncthreads();

    v8f ah;
#pragma unroll
    for (int i = 0; i < 8; ++i) ah[i] = bhv;
#pragma unroll
    for (int c = 0; c < 5; ++c)
      ah = wmma_f16(load_a_frag(xh2, GRU_STRIDE, c * 32),
                    load_b_frag(wfh, c, 8, nt), ah);
#pragma unroll
    for (int i = 0; i < 8; ++i) {
      float hc = tanh_fast(ah[i]);
      h[i] = z[i] * h[i] + (1.f - z[i]) * hc;
    }
    __syncthreads();                 // all xh/xh2 reads of this step done
#pragma unroll
    for (int i = 0; i < 8; ++i)
      xh[(mbase + i) * GRU_STRIDE + C_IN + n] = (_Float16)h[i];
    __syncthreads();                 // h visible before next step's matmuls
  }

#pragma unroll
  for (int i = 0; i < 8; ++i) {
    const int node = tile0 + mbase + i;
    if (node < N) feat[(size_t)node * F1 + C_IN + n] = h[i];
  }
  {
    const int node = tile0 + sr;
    if (node < N) feat[(size_t)node * F1 + sc] = xs[(size_t)node * C_IN + sc];
  }
}

// ---------------------------------------------------------------------------
// SAGE fused matmul: out[node] += [mean_agg | x_dst] @ [Wl; Wr] + bias
// O=128: 8 waves = 8 N-tiles, 1 node-tile/block. O=64: 4 N-tiles, 2 node-tiles.
// Compile-time shape -> fully unrolled WMMA chains, constant LDS addressing.
// ---------------------------------------------------------------------------
template <int F, int O, int cL, int strideH>
__global__ __launch_bounds__(256) void sage_mm_kernel(
    const float* __restrict__ sum, const float* __restrict__ cnt,
    const float* __restrict__ fdst,
    const _Float16* __restrict__ wfl, const _Float16* __restrict__ wfr,
    const float* __restrict__ bias, float* __restrict__ out, int n_dst) {
  __shared__ _Float16 abuf[2 * 16 * SAGE_STRIDE_MAX];

  const int tid = threadIdx.x, lane = tid & 31, w = tid >> 5;
  constexpr int ntiles = O >> 4;           // 8 or 4
  constexpr int npb = 8 / ntiles;          // node tiles per block: 1 or 2
  constexpr int KpadL = cL * 32;
  const int bsub = (npb == 2) ? (w >> 2) : 0;
  const int nt = (npb == 2) ? (w & (ntiles - 1)) : w;

#pragma unroll
  for (int b = 0; b < npb; ++b) {      // stage A = [agg | fdst] as f16
    _Float16* ab = abuf + b * 16 * SAGE_STRIDE_MAX;
    const int tile = blockIdx.x * npb + b;
    for (int idx = tid; idx < 16 * 2 * KpadL; idx += 256) {
      const int r = idx / (2 * KpadL);
      const int c = idx - r * (2 * KpadL);
      int node = tile * 16 + r; if (node >= n_dst) node = n_dst - 1;
      float v;
      if (c < KpadL) {
        if (c < F) {
          float ct = cnt[node];
          v = sum[(size_t)node * F + c] *
              __builtin_amdgcn_rcpf(ct > 1.f ? ct : 1.f);
        } else v = 0.f;
      } else {
        const int k = c - KpadL;
        v = (k < F) ? fdst[(size_t)node * F + k] : 0.f;
      }
      ab[r * strideH + c] = (_Float16)v;
    }
  }
  __syncthreads();

  const _Float16* ab = abuf + bsub * 16 * SAGE_STRIDE_MAX;
  const int tile = blockIdx.x * npb + bsub;
  const int nl = lane & 15, mbase = (lane >> 4) * 8;
  const float bv = bias[nt * 16 + nl];
  v8f acc;
#pragma unroll
  for (int i = 0; i < 8; ++i) acc[i] = bv;
#pragma unroll
  for (int c = 0; c < cL; ++c)
    acc = wmma_f16(load_a_frag(ab, strideH, c * 32),
                   load_b_frag(wfl, c, ntiles, nt), acc);
#pragma unroll
  for (int c = 0; c < cL; ++c)
    acc = wmma_f16(load_a_frag(ab, strideH, KpadL + c * 32),
                   load_b_frag(wfr, c, ntiles, nt), acc);
#pragma unroll
  for (int i = 0; i < 8; ++i) {
    const int node = tile * 16 + mbase + i;
    if (node < n_dst) out[(size_t)node * O + nt * 16 + nl] += acc[i];
  }
}

// ---------------------------------------------------------------------------
// Edge scatter-add (float4 per thread) and degree count.
// unsafeAtomicAdd -> hardware global_atomic_add_f32 (no CAS loop).
// ---------------------------------------------------------------------------
__global__ void scatter_kernel(const int* __restrict__ src, const int* __restrict__ dst,
                               const float* __restrict__ feat, float* __restrict__ sum,
                               int E, int F) {
  const int F4 = F >> 2;
  const long long tid = (long long)blockIdx.x * blockDim.x + threadIdx.x;
  if (tid >= (long long)E * F4) return;
  const int e = (int)(tid / F4);
  const int c = (int)(tid - (long long)e * F4) * 4;
  const int s = src[e], d = dst[e];
  const float* fp = feat + (size_t)s * F + c;
  float* sp = sum + (size_t)d * F + c;
  unsafeAtomicAdd(sp + 0, fp[0]); unsafeAtomicAdd(sp + 1, fp[1]);
  unsafeAtomicAdd(sp + 2, fp[2]); unsafeAtomicAdd(sp + 3, fp[3]);
}

__global__ void count_kernel(const int* __restrict__ dst, float* __restrict__ cnt, int E) {
  const int e = blockIdx.x * blockDim.x + threadIdx.x;
  if (e < E) unsafeAtomicAdd(cnt + dst[e], 1.f);
}

__global__ void zero_kernel(float* __restrict__ p, size_t n) {
  const size_t i = (size_t)blockIdx.x * blockDim.x + threadIdx.x;
  if (i < n) p[i] = 0.f;
}

__global__ void relu_kernel(float* __restrict__ p, size_t n) {
  const size_t i = (size_t)blockIdx.x * blockDim.x + threadIdx.x;
  if (i < n) p[i] = fmaxf(p[i], 0.f);
}

__global__ void pack_kernel(const float* __restrict__ x, float* __restrict__ feat, int N) {
  const size_t i = (size_t)blockIdx.x * blockDim.x + threadIdx.x;
  if (i >= (size_t)N * F1) return;
  const int node = (int)(i / F1), c = (int)(i % F1);
  feat[i] = (c < C_IN) ? x[(size_t)node * C_IN + c] : 0.f;
}

// ---------------------------------------------------------------------------
static inline int cdiv(long long a, long long b) { return (int)((a + b - 1) / b); }

extern "C" void kernel_launch(void* const* d_in, const int* in_sizes, int n_in,
                              void* d_out, int out_size, void* d_ws, size_t ws_size,
                              hipStream_t stream) {
  (void)in_sizes; (void)n_in; (void)out_size; (void)ws_size;

  const float* x_ws  = (const float*)d_in[0];
  const float* x_hru = (const float*)d_in[1];
  const float* x_ch  = (const float*)d_in[2];
  const float* x_gw  = (const float*)d_in[3];
  const float* ts_ws  = (const float*)d_in[4];
  const float* ts_hru = (const float*)d_in[5];
  const float* Wz = (const float*)d_in[6];
  const float* Wr = (const float*)d_in[7];
  const float* Wh = (const float*)d_in[8];
  const float* bz = (const float*)d_in[9];
  const float* br = (const float*)d_in[10];
  const float* bh = (const float*)d_in[11];
  const float* W1l = (const float*)d_in[12];
  const float* b1l = (const float*)d_in[13];
  const float* W1r = (const float*)d_in[14];
  const float* W2l = (const float*)d_in[15];
  const float* b2l = (const float*)d_in[16];
  const float* W2r = (const float*)d_in[17];
  const int* es[7]; const int* ed[7];
  for (int r = 0; r < 7; ++r) {
    es[r] = (const int*)d_in[18 + 2 * r];
    ed[r] = (const int*)d_in[19 + 2 * r];
  }

  // relation tables: r0 ws->hru, r1 hru->gw, r2 hru->ch, r3 ch->gw,
  //                  r4 ch->gw,  r5 hru->ws, r6 gw->ch
  const int Ecnt[7]   = {400000, 600000, 400000, 300000, 100000, 400000, 300000};
  const int srcOff[7] = {OFF_WS, OFF_HRU, OFF_HRU, OFF_CH, OFF_CH, OFF_HRU, OFF_GW};
  const int dstOff[7] = {OFF_HRU, OFF_GW, OFF_CH, OFF_GW, OFF_GW, OFF_WS, OFF_CH};
  const int dstN[7]   = {N_HRU, N_GW, N_CH, N_GW, N_GW, N_WS, N_CH};

  // ---- workspace carve-up -------------------------------------------------
  char* w = (char*)d_ws;
  float* feat = (float*)w;  w += (size_t)N_TOT * F1 * 4;       // [x|h] per node
  float* h1   = (float*)w;  w += (size_t)N_TOT * HID * 4;      // layer1 output
  float* sum  = (float*)w;  w += (size_t)N_GW * F1 * 4;        // per-rel msg sum
  float* cnt  = (float*)w;  w += (size_t)N_GW * 4;             // per-rel degree
  const int FR1 = 5 * 8 * 32 * 16;   // 160x128 frag halves = 20480
  const int FR2 = 4 * 4 * 32 * 16;   // 128x64  frag halves = 8192
  _Float16* wfz = (_Float16*)w; w += FR1 * 2;
  _Float16* wfr = (_Float16*)w; w += FR1 * 2;
  _Float16* wfh = (_Float16*)w; w += FR1 * 2;
  _Float16 *wf1l[7], *wf1r[7], *wf2l[7], *wf2r[7];
  for (int r = 0; r < 7; ++r) { wf1l[r] = (_Float16*)w; w += FR1 * 2;
                                wf1r[r] = (_Float16*)w; w += FR1 * 2; }
  for (int r = 0; r < 7; ++r) { wf2l[r] = (_Float16*)w; w += FR2 * 2;
                                wf2r[r] = (_Float16*)w; w += FR2 * 2; }

  // ---- weight pre-swizzle to f16 B-fragments ------------------------------
  const int pb1 = cdiv(FR1, 256), pb2 = cdiv(FR2, 256);
  wprep_kernel<<<pb1, 256, 0, stream>>>(Wz, wfz, F1, HID, 5);
  wprep_kernel<<<pb1, 256, 0, stream>>>(Wr, wfr, F1, HID, 5);
  wprep_kernel<<<pb1, 256, 0, stream>>>(Wh, wfh, F1, HID, 5);
  for (int r = 0; r < 7; ++r) {
    wprep_kernel<<<pb1, 256, 0, stream>>>(W1l + (size_t)r * F1 * HID, wf1l[r], F1, HID, 5);
    wprep_kernel<<<pb1, 256, 0, stream>>>(W1r + (size_t)r * F1 * HID, wf1r[r], F1, HID, 5);
    wprep_kernel<<<pb2, 256, 0, stream>>>(W2l + (size_t)r * HID * OUTD, wf2l[r], HID, OUTD, 4);
    wprep_kernel<<<pb2, 256, 0, stream>>>(W2r + (size_t)r * HID * OUTD, wf2r[r], HID, OUTD, 4);
  }

  // ---- node features: [x|h] for ws/hru (GRU), [x|0] for ch/gw -------------
  pack_kernel<<<cdiv((long long)N_CH * F1, 256), 256, 0, stream>>>(x_ch, feat + (size_t)OFF_CH * F1, N_CH);
  pack_kernel<<<cdiv((long long)N_GW * F1, 256), 256, 0, stream>>>(x_gw, feat + (size_t)OFF_GW * F1, N_GW);
  gru_kernel<<<cdiv(N_WS, 16), 256, 0, stream>>>(ts_ws, x_ws, wfz, wfr, wfh, bz, br, bh,
                                                 feat + (size_t)OFF_WS * F1, N_WS);
  gru_kernel<<<cdiv(N_HRU, 16), 256, 0, stream>>>(ts_hru, x_hru, wfz, wfr, wfh, bz, br, bh,
                                                  feat + (size_t)OFF_HRU * F1, N_HRU);

  // ---- hetero layer 1 (F=144 -> 128) + ReLU -------------------------------
  zero_kernel<<<cdiv((long long)N_TOT * HID, 256), 256, 0, stream>>>(h1, (size_t)N_TOT * HID);
  for (int r = 0; r < 7; ++r) {
    const int nd = dstN[r], E = Ecnt[r];
    zero_kernel<<<cdiv((long long)nd * F1, 256), 256, 0, stream>>>(sum, (size_t)nd * F1);
    zero_kernel<<<cdiv(nd, 256), 256, 0, stream>>>(cnt, (size_t)nd);
    scatter_kernel<<<cdiv((long long)E * (F1 / 4), 256), 256, 0, stream>>>(
        es[r], ed[r], feat + (size_t)srcOff[r] * F1, sum, E, F1);
    count_kernel<<<cdiv(E, 256), 256, 0, stream>>>(ed[r], cnt, E);
    sage_mm_kernel<F1, HID, 5, 424><<<cdiv(nd, 16), 256, 0, stream>>>(
        sum, cnt, feat + (size_t)dstOff[r] * F1, wf1l[r], wf1r[r],
        b1l + (size_t)r * HID, h1 + (size_t)dstOff[r] * HID, nd);
  }
  relu_kernel<<<cdiv((long long)N_TOT * HID, 256), 256, 0, stream>>>(h1, (size_t)N_TOT * HID);

  // ---- hetero layer 2 (F=128 -> 64) into d_out ----------------------------
  float* out = (float*)d_out;
  zero_kernel<<<cdiv((long long)N_TOT * OUTD, 256), 256, 0, stream>>>(out, (size_t)N_TOT * OUTD);
  for (int r = 0; r < 7; ++r) {
    const int nd = dstN[r], E = Ecnt[r];
    zero_kernel<<<cdiv((long long)nd * HID, 256), 256, 0, stream>>>(sum, (size_t)nd * HID);
    zero_kernel<<<cdiv(nd, 256), 256, 0, stream>>>(cnt, (size_t)nd);
    scatter_kernel<<<cdiv((long long)E * (HID / 4), 256), 256, 0, stream>>>(
        es[r], ed[r], h1 + (size_t)srcOff[r] * HID, sum, E, HID);
    count_kernel<<<cdiv(E, 256), 256, 0, stream>>>(ed[r], cnt, E);
    sage_mm_kernel<HID, OUTD, 4, 296><<<cdiv(nd, 32), 256, 0, stream>>>(
        sum, cnt, h1 + (size_t)dstOff[r] * HID, wf2l[r], wf2r[r],
        b2l + (size_t)r * OUTD, out + (size_t)dstOff[r] * OUTD, nd);
  }
}
